// BiLSTMCRF_73667279061708
// MI455X (gfx1250) — matrix operand order repaired and checked by reading, not compiled, not run on hardware
//
#include <hip/hip_runtime.h>
#include <hip/hip_bf16.h>

typedef _Float16 half_t;
typedef __attribute__((ext_vector_type(16))) _Float16 v16h;
typedef __attribute__((ext_vector_type(8)))  _Float16 v8h;
typedef __attribute__((ext_vector_type(8)))  float    v8f;

#define TT 512
#define BB 64
#define EE 256
#define HH 200
#define CT 20
#define KP 224      // padded hidden K for recurrence (200 -> 224 = 7*32)
#define HP 208      // padded hidden N per gate (200 -> 208 = 13*16)
#define NHT 13
#define NT2 52      // 4 gates * 13 hidden tiles
#define SCAN_WAVES 26
#define FCK 416     // padded 2H (400 -> 416 = 13*32)
#define NEG_BIG (-1e30f)

union AFrag { v16h v; v8h h[2]; };

// A fragment: 16x32 f16, row-major source with leading dim ldk (halves).
// lanes 0-15: M=lane, elems 0..7 = K(k0+0..7), elems 8..15 = K(k0+16..23)
// lanes 16-31: M=lane-16, elems 0..7 = K(k0+8..15), elems 8..15 = K(k0+24..31)
__device__ __forceinline__ v16h load_a16(const half_t* base, int ldk, int k0, int lane) {
  int m = lane & 15;
  int koff = (lane < 16) ? 0 : 8;
  const half_t* p = base + (long)m * ldk + k0 + koff;
  AFrag u;
  u.h[0] = *(const v8h*)(p);
  u.h[1] = *(const v8h*)(p + 16);
  return u.v;
}

// B fragment: 32x16 f16 from B^T stored row-major [N][K] with leading dim ldk.
// lanes 0-15: N=lane, K = k0+0..7,16..23 ; lanes 16-31: N=lane-16, K = k0+8..15,24..31
__device__ __forceinline__ v16h load_b16(const half_t* base, int ldk, int k0, int lane) {
  int n = lane & 15;
  int koff = (lane < 16) ? 0 : 8;
  const half_t* p = base + (long)n * ldk + k0 + koff;
  AFrag u;
  u.h[0] = *(const v8h*)(p);
  u.h[1] = *(const v8h*)(p + 16);
  return u.v;
}

__device__ __forceinline__ v8f wmma16(v16h a, v16h b, v8f c) {
  return __builtin_amdgcn_wmma_f32_16x16x32_f16(false, a, false, b, (short)0, c, false, false);
}

__device__ __forceinline__ float sigmoidf_(float x) { return 1.f / (1.f + __expf(-x)); }

// ---------------- embedding + layernorm -> f16, row = t*B + b ----------------
__global__ __launch_bounds__(256) void k_embed_ln(
    const int* __restrict__ words, const float* __restrict__ etab,
    const float* __restrict__ gamma, const float* __restrict__ beta,
    half_t* __restrict__ x16) {
  int u = (int)((blockIdx.x * blockDim.x + threadIdx.x) >> 5);
  int lane = threadIdx.x & 31;
  if (u >= TT * BB) return;
  int t = u / BB, b = u % BB;
  int w = words[b * TT + t];
  const float* e = etab + (long)w * EE;
  float x[8]; float s = 0.f;
  #pragma unroll
  for (int j = 0; j < 8; j++) { x[j] = e[lane + 32 * j]; s += x[j]; }
  #pragma unroll
  for (int off = 16; off >= 1; off >>= 1) s += __shfl_xor(s, off, 32);
  float mu = s * (1.f / EE);
  float v = 0.f;
  #pragma unroll
  for (int j = 0; j < 8; j++) { float d = x[j] - mu; v += d * d; }
  #pragma unroll
  for (int off = 16; off >= 1; off >>= 1) v += __shfl_xor(v, off, 32);
  float rs = rsqrtf(v * (1.f / EE) + 1e-5f);
  #pragma unroll
  for (int j = 0; j < 8; j++) {
    int ei = lane + 32 * j;
    float y = (x[j] - mu) * rs * gamma[ei] + beta[ei];
    x16[(long)u * EE + ei] = (half_t)y;
  }
}

// ---------------- helpers: zero / pad ----------------
__global__ void k_zero_half(half_t* __restrict__ p, long n) {
  long i = (long)blockIdx.x * blockDim.x + threadIdx.x;
  long stride = (long)gridDim.x * blockDim.x;
  for (; i < n; i += stride) p[i] = (half_t)0.f;
}

// wihp[dir][gate][hc(208)][k(256)] = w_ih[dir][gate*200+hc][k] (zero padded rows)
__global__ void k_pad_wih(const float* __restrict__ wif, const float* __restrict__ wib,
                          half_t* __restrict__ dst) {
  int idx = blockIdx.x * blockDim.x + threadIdx.x;
  const int n = 2 * 4 * HP * EE;
  if (idx >= n) return;
  int k = idx % EE;
  int hc = (idx / EE) % HP;
  int gate = (idx / (EE * HP)) % 4;
  int dir = idx / (EE * HP * 4);
  const float* src = dir ? wib : wif;
  float v = (hc < HH) ? src[(gate * HH + hc) * EE + k] : 0.f;
  dst[idx] = (half_t)v;
}

// whhp[dir][gate][hc(208)][k(224)] = w_hh[dir][gate*200+hc][k] (zero padded)
__global__ void k_pad_whh(const float* __restrict__ whf, const float* __restrict__ whb,
                          half_t* __restrict__ dst) {
  int idx = blockIdx.x * blockDim.x + threadIdx.x;
  const int n = 2 * 4 * HP * KP;
  if (idx >= n) return;
  int k = idx % KP;
  int hc = (idx / KP) % HP;
  int gate = (idx / (KP * HP)) % 4;
  int dir = idx / (KP * HP * 4);
  const float* src = dir ? whb : whf;
  float v = 0.f;
  if (hc < HH && k < HH) v = src[(gate * HH + hc) * HH + k];
  dst[idx] = (half_t)v;
}

// fcwp[32][416] = fc_w[20][400] zero padded
__global__ void k_pad_fcw(const float* __restrict__ fcw, half_t* __restrict__ dst) {
  int idx = blockIdx.x * blockDim.x + threadIdx.x;
  const int n = 32 * FCK;
  if (idx >= n) return;
  int k = idx % FCK;
  int r = idx / FCK;
  float v = (r < CT && k < 2 * HH) ? fcw[r * (2 * HH) + k] : 0.f;
  dst[idx] = (half_t)v;
}

// ---------------- input projection GEMM: gates = x @ w_ih^T + b ----------------
// M = T*B (row = t*B+b), K = 256, N padded to 4x208; fragment-native output:
// gates[dir][(mt*52 + nt2)*32 + lane][8] f32, mt = t*4 + mb, nt2 = gate*13 + nh.
__global__ __launch_bounds__(128) void k_gemm_gates(
    const half_t* __restrict__ x16, const half_t* __restrict__ wihp,
    const float* __restrict__ bias_f, const float* __restrict__ bias_b,
    float* __restrict__ gates_f, float* __restrict__ gates_b) {
  int lane = threadIdx.x & 31;
  int wave = threadIdx.x >> 5;
  int mt = blockIdx.x;                 // 0..2047
  int nd = blockIdx.y * 4 + wave;      // 0..103
  int dir = (nd >= NT2);
  int nt2 = dir ? nd - NT2 : nd;
  int gate = nt2 / NHT, nh = nt2 % NHT;
  const half_t* w = wihp + ((long)dir * 4 * HP + gate * HP + nh * 16) * EE;
  const float* bias = dir ? bias_b : bias_f;
  float* out = dir ? gates_b : gates_f;
  int m0 = mt * 16;
  v8f acc;
  #pragma unroll
  for (int i = 0; i < 8; i++) acc[i] = 0.f;
  #pragma unroll
  for (int kt = 0; kt < 8; kt++) {
    v16h a = load_a16(x16 + (long)m0 * EE, EE, kt * 32, lane);
    v16h b = load_b16(w, EE, kt * 32, lane);
    acc = wmma16(a, b, acc);
  }
  int hcol = nh * 16 + (lane & 15);
  float bv = (hcol < HH) ? bias[gate * HH + hcol] : 0.f;
  #pragma unroll
  for (int i = 0; i < 8; i++) acc[i] += bv;
  *(v8f*)(out + (((long)mt * NT2 + nt2) * 32 + lane) * 8) = acc;
}

// ---------------- LSTM scan: one workgroup per direction, WMMA recurrence -----
__global__ __launch_bounds__(832) void k_lstm_scan(
    const float* __restrict__ gates_f, const float* __restrict__ gates_b,
    const half_t* __restrict__ whhp, const int* __restrict__ seq_len,
    half_t* __restrict__ hcat) {
  int dir = blockIdx.x;
  const float* g = dir ? gates_b : gates_f;
  const half_t* wh = whhp + (long)dir * (4 * HP * KP);
  __shared__ half_t hbuf[2][BB][KP];
  for (int i = threadIdx.x; i < 2 * BB * KP; i += blockDim.x)
    ((half_t*)hbuf)[i] = (half_t)0.f;
  __syncthreads();

  int lane = threadIdx.x & 31;
  int wave = threadIdx.x >> 5;

  float c[2][8], h[2][8];
  int pm[2], pn[2], slen[2][8];
  #pragma unroll
  for (int q = 0; q < 2; q++) {
    int p = wave + q * SCAN_WAVES;      // 0..51
    pm[q] = p / NHT;                    // batch tile 0..3
    pn[q] = p % NHT;                    // hidden tile 0..12
    #pragma unroll
    for (int r = 0; r < 8; r++) {
      c[q][r] = 0.f; h[q][r] = 0.f;
      int b = pm[q] * 16 + r + ((lane >= 16) ? 8 : 0);
      slen[q][r] = seq_len[b];
    }
  }

  for (int s = 0; s < TT; s++) {
    int t = dir ? (TT - 1 - s) : s;
    int tn = dir ? (t - 1) : (t + 1);
    int rd = s & 1, wr = rd ^ 1;
    #pragma unroll
    for (int q = 0; q < 2; q++) {
      int m0 = pm[q] * 16, nh0 = pn[q] * 16;
      int hcol = nh0 + (lane & 15);
      v8f acc[4];
      // fragment-native, fully coalesced C-tile loads (one v8f per gate)
      #pragma unroll
      for (int gi = 0; gi < 4; gi++) {
        acc[gi] = *(const v8f*)(g +
            (((long)(t * 4 + pm[q]) * NT2 + gi * NHT + pn[q]) * 32 + lane) * 8);
      }
      // prefetch next step's fragments while WMMAs run (global_prefetch_b8)
      if (tn >= 0 && tn < TT) {
        #pragma unroll
        for (int gi = 0; gi < 4; gi++) {
          __builtin_prefetch(g +
              (((long)(tn * 4 + pm[q]) * NT2 + gi * NHT + pn[q]) * 32 + lane) * 8, 0, 3);
        }
      }
      #pragma unroll
      for (int kt = 0; kt < 7; kt++) {
        v16h af = load_a16(&hbuf[rd][m0][0], KP, kt * 32, lane);
        #pragma unroll
        for (int gi = 0; gi < 4; gi++) {
          v16h bf = load_b16(wh + (long)(gi * HP + nh0) * KP, KP, kt * 32, lane);
          acc[gi] = wmma16(af, bf, acc[gi]);
        }
      }
      #pragma unroll
      for (int r = 0; r < 8; r++) {
        float ig = sigmoidf_(acc[0][r]);
        float fg = sigmoidf_(acc[1][r]);
        float gg = tanhf(acc[2][r]);
        float og = sigmoidf_(acc[3][r]);
        float cn = fg * c[q][r] + ig * gg;
        float hn = og * tanhf(cn);
        float mt = (t < slen[q][r]) ? 1.f : 0.f;
        c[q][r] = mt * cn + (1.f - mt) * c[q][r];
        h[q][r] = mt * hn + (1.f - mt) * h[q][r];
      }
      #pragma unroll
      for (int r = 0; r < 8; r++) {
        int b = m0 + r + ((lane >= 16) ? 8 : 0);
        if (hcol < HH) {
          half_t hv = (half_t)h[q][r];
          hbuf[wr][b][hcol] = hv;
          hcat[((long)b * TT + t) * FCK + dir * HH + hcol] = hv;
        } else {
          hbuf[wr][b][hcol] = (half_t)0.f;   // keep K-padding zero
        }
      }
    }
    __syncthreads();
  }
}

// ---------------- FC + log_softmax, fused WMMA ----------------
__global__ __launch_bounds__(128) void k_fc_lsm(
    const half_t* __restrict__ hcat, const half_t* __restrict__ fcwp,
    const float* __restrict__ fcb, float* __restrict__ logits) {
  int lane = threadIdx.x & 31;
  int wave = threadIdx.x >> 5;
  int mt = blockIdx.x * 4 + wave;     // 0..2047
  int m0 = mt * 16;
  v8f a0, a1;
  #pragma unroll
  for (int i = 0; i < 8; i++) { a0[i] = 0.f; a1[i] = 0.f; }
  #pragma unroll
  for (int kt = 0; kt < 13; kt++) {
    v16h a = load_a16(hcat + (long)m0 * FCK, FCK, kt * 32, lane);
    v16h b0 = load_b16(fcwp, FCK, kt * 32, lane);
    v16h b1 = load_b16(fcwp + 16 * FCK, FCK, kt * 32, lane);
    a0 = wmma16(a, b0, a0);
    a1 = wmma16(a, b1, a1);
  }
  int c0 = lane & 15, c1 = 16 + c0;
  float b0v = fcb[c0];
  float b1v = (c1 < CT) ? fcb[c1] : 0.f;
  #pragma unroll
  for (int r = 0; r < 8; r++) {
    float v0 = a0[r] + b0v;
    float v1 = (c1 < CT) ? (a1[r] + b1v) : NEG_BIG;
    float m = fmaxf(v0, v1);
    #pragma unroll
    for (int off = 8; off >= 1; off >>= 1) m = fmaxf(m, __shfl_xor(m, off, 32));
    float s = __expf(v0 - m) + ((c1 < CT) ? __expf(v1 - m) : 0.f);
    #pragma unroll
    for (int off = 8; off >= 1; off >>= 1) s += __shfl_xor(s, off, 32);
    float lse = m + __logf(s);
    int row = m0 + r + ((lane >= 16) ? 8 : 0);
    logits[(long)row * CT + c0] = v0 - lse;
    if (c1 < CT) logits[(long)row * CT + c1] = v1 - lse;
  }
}

// ---------------- CRF NLL per sample: one wave per b ----------------
__global__ __launch_bounds__(32) void k_crf(
    const float* __restrict__ logits, const int* __restrict__ seq_len,
    const int* __restrict__ target, const float* __restrict__ trans,
    const float* __restrict__ startv, const float* __restrict__ endv,
    float* __restrict__ lossb) {
  int b = blockIdx.x;
  int lane = threadIdx.x;
  __shared__ float sTrans[CT * CT];
  __shared__ float sAlpha[CT];
  for (int i = lane; i < CT * CT; i += 32) sTrans[i] = trans[i];
  int L = seq_len[b];
  const float* lg = logits + (long)b * TT * CT;
  int cl = (lane < CT) ? lane : (CT - 1);
  if (lane < CT) sAlpha[lane] = startv[lane] + lg[lane];
  __syncthreads();
  for (int t = 1; t < TT; t++) {
    if (t < L) {
      float emit = (lane < CT) ? lg[t * CT + lane] : 0.f;
      float m = NEG_BIG;
      #pragma unroll 4
      for (int i = 0; i < CT; i++) m = fmaxf(m, sAlpha[i] + sTrans[i * CT + cl]);
      float ssum = 0.f;
      #pragma unroll 4
      for (int i = 0; i < CT; i++) ssum += __expf(sAlpha[i] + sTrans[i * CT + cl] - m);
      float na = emit + m + __logf(ssum);
      __syncthreads();
      if (lane < CT) sAlpha[lane] = na;
      __syncthreads();
    }
  }
  float fin = (lane < CT) ? (sAlpha[lane] + endv[lane]) : NEG_BIG;
  float m = fin;
  #pragma unroll
  for (int off = 16; off >= 1; off >>= 1) m = fmaxf(m, __shfl_xor(m, off, 32));
  float e = (lane < CT) ? __expf(fin - m) : 0.f;
  #pragma unroll
  for (int off = 16; off >= 1; off >>= 1) e += __shfl_xor(e, off, 32);
  float logZ = m + __logf(e);
  const int* tg = target + b * TT;
  float acc = 0.f;
  for (int t = lane; t < TT; t += 32) {
    float mt = (t < L) ? 1.f : 0.f;
    acc += mt * lg[t * CT + tg[t]];
    if (t >= 1) acc += mt * trans[tg[t - 1] * CT + tg[t]];
  }
  #pragma unroll
  for (int off = 16; off >= 1; off >>= 1) acc += __shfl_xor(acc, off, 32);
  if (lane == 0) {
    float gold = acc + startv[tg[0]] + endv[tg[L - 1]];
    lossb[b] = logZ - gold;
  }
}

__global__ __launch_bounds__(32) void k_reduce_loss(const float* __restrict__ lossb,
                                                    float* __restrict__ out) {
  int lane = threadIdx.x;
  float a = lossb[lane] + lossb[lane + 32];
  #pragma unroll
  for (int off = 16; off >= 1; off >>= 1) a += __shfl_xor(a, off, 32);
  if (lane == 0) out[0] = a * (1.f / BB);
}

// ---------------- launcher ----------------
extern "C" void kernel_launch(void* const* d_in, const int* in_sizes, int n_in,
                              void* d_out, int out_size, void* d_ws, size_t ws_size,
                              hipStream_t stream) {
  (void)in_sizes; (void)n_in; (void)out_size; (void)ws_size;
  const int*   words    = (const int*)  d_in[0];
  const int*   seq_len  = (const int*)  d_in[1];
  const int*   target   = (const int*)  d_in[2];
  const float* etab     = (const float*)d_in[3];
  const float* ln_gamma = (const float*)d_in[4];
  const float* ln_beta  = (const float*)d_in[5];
  const float* w_ih_f   = (const float*)d_in[6];
  const float* w_hh_f   = (const float*)d_in[7];
  const float* b_f      = (const float*)d_in[8];
  const float* w_ih_b   = (const float*)d_in[9];
  const float* w_hh_b   = (const float*)d_in[10];
  const float* b_b      = (const float*)d_in[11];
  const float* fc_w     = (const float*)d_in[12];
  const float* fc_b     = (const float*)d_in[13];
  const float* trans    = (const float*)d_in[14];
  const float* startv   = (const float*)d_in[15];
  const float* endv     = (const float*)d_in[16];
  float* out = (float*)d_out;

  char* ws = (char*)d_ws;
  size_t off = 0;
  auto carve = [&](size_t bytes) {
    void* p = ws + off;
    off = (off + bytes + 255) & ~(size_t)255;
    return p;
  };
  const size_t gate_elems = (size_t)2048 * NT2 * 32 * 8;             // per direction
  half_t* x16     = (half_t*)carve((size_t)TT * BB * EE * 2);        // 16.8 MB
  float*  gates_f = (float*) carve(gate_elems * 4);                  // 109 MB
  float*  gates_b = (float*) carve(gate_elems * 4);                  // 109 MB
  half_t* wihp    = (half_t*)carve((size_t)2 * 4 * HP * EE * 2);     // 0.85 MB
  half_t* whhp    = (half_t*)carve((size_t)2 * 4 * HP * KP * 2);     // 0.75 MB
  half_t* fcwp    = (half_t*)carve((size_t)32 * FCK * 2);
  half_t* hcat    = (half_t*)carve((size_t)BB * TT * FCK * 2);       // 27.3 MB
  float*  logits  = (float*) carve((size_t)BB * TT * CT * 4);        // 2.6 MB
  float*  lossb   = (float*) carve((size_t)BB * 4);

  // 1) embedding + layernorm -> f16 (row = t*B + b)
  k_embed_ln<<<dim3((TT * BB) / 8), dim3(256), 0, stream>>>(words, etab, ln_gamma, ln_beta, x16);
  // 2) zero feats buffer (padding cols must be 0 for FC K-padding)
  k_zero_half<<<dim3(8192), dim3(256), 0, stream>>>(hcat, (long)BB * TT * FCK);
  // 3) padded f16 weight layouts
  k_pad_wih<<<dim3((2 * 4 * HP * EE + 255) / 256), dim3(256), 0, stream>>>(w_ih_f, w_ih_b, wihp);
  k_pad_whh<<<dim3((2 * 4 * HP * KP + 255) / 256), dim3(256), 0, stream>>>(w_hh_f, w_hh_b, whhp);
  k_pad_fcw<<<dim3((32 * FCK + 255) / 256), dim3(256), 0, stream>>>(fc_w, fcwp);
  // 4) input-projection GEMM (WMMA), fragment-native gate output, both directions
  k_gemm_gates<<<dim3(2048, 26), dim3(128), 0, stream>>>(x16, wihp, b_f, b_b, gates_f, gates_b);
  // 5) sequential LSTM scans (WMMA recurrence), one WGP per direction
  k_lstm_scan<<<dim3(2), dim3(832), 0, stream>>>(gates_f, gates_b, whhp, seq_len, hcat);
  // 6) FC + log_softmax (WMMA)
  k_fc_lsm<<<dim3(512), dim3(128), 0, stream>>>(hcat, fcwp, fc_b, logits);
  // 7) CRF NLL per sample
  k_crf<<<dim3(BB), dim3(32), 0, stream>>>(logits, seq_len, target, trans, startv, endv, lossb);
  // 8) mean
  k_reduce_loss<<<dim3(1), dim3(32), 0, stream>>>(lossb, out);
}
